// NPInstantaneousTransitionPrior_33998961115148
// MI455X (gfx1250) — matrix operand order repaired and checked by reading, not compiled, not run on hardware
//
#include <hip/hip_runtime.h>
#include <math.h>

// ---------------- CDNA5 WMMA vector types ----------------
typedef __attribute__((ext_vector_type(16))) _Float16 v16h;
typedef __attribute__((ext_vector_type(8)))  _Float16 v8h;
typedef __attribute__((ext_vector_type(8)))  float    v8f;

#define NSAMP  (256 * 512)     // B * (T - LAGS)
#define TLEN   513
#define DLAT   7
#define SLOPE  0.2f
#define ITERS  4               // super-tiles per block (block = 4 waves x 16 samples)

// ---- A fragment: 16x32 f16 from row-major h[16][64], K base kb ----
// lanes 0-15: M=lane, K = kb+{0..7} (v0-3) and kb+16+{0..7} (v4-7)
// lanes 16-31: M=lane-16, K = kb+8+{0..7} and kb+24+{0..7}
__device__ __forceinline__ v16h ldA(const _Float16* h, int lane, int kb) {
    int m  = lane & 15;
    int hf = lane >> 4;
    v8h lo = *(const v8h*)(h + m * 64 + kb + hf * 8);
    v8h hi = *(const v8h*)(h + m * 64 + kb + 16 + hf * 8);
    v16h a;
#pragma unroll
    for (int i = 0; i < 8; ++i) { a[i] = lo[i]; a[8 + i] = hi[i]; }
    return a;
}

// ---- B fragment: B[k][n] = wgt[n*64 + k]  (contiguous in k per lane) ----
// lane: n = nt*16 + lane%16 ; K = kb + (lane/16)*8 + {0..7} and kb + 16 + (lane/16)*8 + {0..7}
__device__ __forceinline__ v16h ldB(const _Float16* wgt, int lane, int nt, int kb) {
    int n  = nt * 16 + (lane & 15);
    int hf = lane >> 4;
    v8h lo = *(const v8h*)(wgt + n * 64 + kb + hf * 8);
    v8h hi = *(const v8h*)(wgt + n * 64 + kb + 16 + hf * 8);
    v16h b;
#pragma unroll
    for (int i = 0; i < 8; ++i) { b[i] = lo[i]; b[8 + i] = hi[i]; }
    return b;
}

__device__ __forceinline__ v8f wmma_f16(v16h a, v16h b, v8f c) {
    return __builtin_amdgcn_wmma_f32_16x16x32_f16(false, a, false, b, (short)0, c,
                                                  false, false);
}

__global__ void zero_logdet_kernel(float* p) {
    int i = blockIdx.x * blockDim.x + threadIdx.x;
    if (i < 256) p[i] = 0.0f;
}

// One launch per MLP. Block = 128 threads (4 wave32); each wave does ITERS
// 16-sample tiles; block covers 256 consecutive samples (within one batch row).
__global__ void __launch_bounds__(128)
prior_mlp_kernel(const float* __restrict__ x,
                 const float* __restrict__ W1, const float* __restrict__ B1,
                 const float* __restrict__ W2, const float* __restrict__ B2,
                 const float* __restrict__ W3, const float* __restrict__ B3,
                 const float* __restrict__ Wo, const float* __restrict__ Bo,
                 float* __restrict__ res_out,    // [N,7], column `mlp`
                 float* __restrict__ logdet_out, // [256], atomic accumulate
                 float* __restrict__ hist_out,   // [N, fin-1] for this mlp
                 int fin, int n_yy, int yy_off, int n_xp, int xp_off,
                 int self_col, int mlp) {
    // ---------------- LDS ----------------
    __shared__ __align__(32) _Float16 s_w2 [64 * 64];  // W2[c][j]   (fwd B)
    __shared__ __align__(32) _Float16 s_w3 [64 * 64];  // W3[c][j]   (fwd B)
    __shared__ __align__(32) _Float16 s_w2t[64 * 64];  // W2[c][j] -> [j][c] (bwd B)
    __shared__ __align__(32) _Float16 s_w3t[64 * 64];  // W3 transposed      (bwd B)
    __shared__ __align__(32) _Float16 s_w1t[16 * 64];  // W1^T zero-padded   (J B)
    __shared__ __align__(16) float    s_w1 [64 * 8];   // W1[c][j], stride 8, 0-pad
    __shared__ float s_b1[64], s_b2[64], s_b3[64], s_wo[64];
    __shared__ float s_bo;
    __shared__ float s_in[4][16 * 8];                  // per-wave inputs
    __shared__ __align__(32) _Float16 s_h[4][16 * 64]; // h / g staging
    __shared__ unsigned char s_d1[4][16 * 64];         // leaky mask flags
    __shared__ unsigned char s_d2[4][16 * 64];
    __shared__ unsigned char s_d3[4][16 * 64];

    const int tid  = threadIdx.x;
    const int w    = tid >> 5;
    const int lane = tid & 31;
    const int nl   = lane & 15;   // n_local / sample-in-tile
    const int hf   = lane >> 4;   // half-wave id

    // ---------------- preload weights (block-cooperative) ----------------
    for (int k = tid; k < 64 * 64; k += 128) {
        int c = k >> 6, j = k & 63;
        _Float16 w2v = (_Float16)W2[k];
        _Float16 w3v = (_Float16)W3[k];
        s_w2[k] = w2v;  s_w2t[j * 64 + c] = w2v;
        s_w3[k] = w3v;  s_w3t[j * 64 + c] = w3v;
    }
    for (int k = tid; k < 64 * 8; k += 128) {
        int c = k >> 3, j = k & 7;
        s_w1[k] = (j < fin) ? W1[c * fin + j] : 0.0f;
    }
    for (int k = tid; k < 16 * 64; k += 128) {
        int n = k >> 6, c = k & 63;               // B[k=c][n=j] = W1[c][n]
        s_w1t[k] = (n < fin) ? (_Float16)W1[c * fin + n] : (_Float16)0.0f;
    }
    if (tid < 64) {
        s_b1[tid] = B1[tid]; s_b2[tid] = B2[tid];
        s_b3[tid] = B3[tid]; s_wo[tid] = Wo[tid];
    }
    if (tid == 0) s_bo = Bo[0];
    __syncthreads();

    const int base = blockIdx.x * (ITERS * 64);
    float logdet_part = 0.0f;

    for (int it = 0; it < ITERS; ++it) {
        const int tile = base + it * 64 + w * 16;  // first sample of this wave

        // ---------------- gather inputs: [yy | x_prev | x_self] ----------
        {
            int m  = nl;
            int s  = tile + m;
            int bb = s >> 9, tt = s & 511;
            const float* row0 = x + ((size_t)bb * TLEN + tt) * DLAT; // lagged
            const float* row1 = row0 + DLAT;                         // current
            int j0 = hf * 4;
            int j1 = (fin < j0 + 4) ? fin : (j0 + 4);
            for (int j = j0; j < j1; ++j) {
                float v;
                if (j < n_yy)             v = row0[yy_off + j];
                else if (j < n_yy + n_xp) v = row1[xp_off + (j - n_yy)];
                else                      v = row1[self_col];
                s_in[w][m * 8 + j] = v;
            }
        }

        // ---------------- layer 1 (fin -> 64), VALU f32 ----------------
        {
            int m = nl;
            int c0 = hf * 32;
            for (int t = 0; t < 32; ++t) {
                int c = c0 + t;
                float pre = s_b1[c];
                for (int j = 0; j < fin; ++j)
                    pre += s_in[w][m * 8 + j] * s_w1[c * 8 + j];
                float dd = pre > 0.0f ? 1.0f : SLOPE;
                s_h [w][m * 64 + c] = (_Float16)(pre * dd); // leaky relu
                s_d1[w][m * 64 + c] = pre > 0.0f ? 1 : 0;
            }
        }

        // ---------------- layer 2 fwd: h2 = lrelu(h1 @ W2^T + b2) --------
        {
            v8f acc[4];
#pragma unroll
            for (int nt = 0; nt < 4; ++nt) {
                v8f c = {};
                for (int kb = 0; kb < 64; kb += 32)
                    c = wmma_f16(ldA(s_h[w], lane, kb), ldB(s_w2, lane, nt, kb), c);
                acc[nt] = c;
            }
#pragma unroll
            for (int nt = 0; nt < 4; ++nt)
#pragma unroll
                for (int r = 0; r < 8; ++r) {
                    int m = r + hf * 8, c = nt * 16 + nl;
                    float pre = acc[nt][r] + s_b2[c];
                    float dd  = pre > 0.0f ? 1.0f : SLOPE;
                    s_h [w][m * 64 + c] = (_Float16)(pre * dd);
                    s_d2[w][m * 64 + c] = pre > 0.0f ? 1 : 0;
                }
        }

        // ---------------- layer 3 fwd ----------------
        {
            v8f acc[4];
#pragma unroll
            for (int nt = 0; nt < 4; ++nt) {
                v8f c = {};
                for (int kb = 0; kb < 64; kb += 32)
                    c = wmma_f16(ldA(s_h[w], lane, kb), ldB(s_w3, lane, nt, kb), c);
                acc[nt] = c;
            }
#pragma unroll
            for (int nt = 0; nt < 4; ++nt)
#pragma unroll
                for (int r = 0; r < 8; ++r) {
                    int m = r + hf * 8, c = nt * 16 + nl;
                    float pre = acc[nt][r] + s_b3[c];
                    float dd  = pre > 0.0f ? 1.0f : SLOPE;
                    s_h [w][m * 64 + c] = (_Float16)(pre * dd);
                    s_d3[w][m * 64 + c] = pre > 0.0f ? 1 : 0;
                }
        }

        // ---------------- output head: res = h3 . Wo + bo ----------------
        if (hf == 0) {
            int m = nl;
            float acc = s_bo;
            const v8h* hp = (const v8h*)(s_h[w] + m * 64);
#pragma unroll
            for (int ch = 0; ch < 8; ++ch) {
                v8h hv = hp[ch];
#pragma unroll
                for (int e = 0; e < 8; ++e)
                    acc += (float)hv[e] * s_wo[ch * 8 + e];
            }
            res_out[(size_t)(tile + m) * DLAT + mlp] = acc;
        }

        // ---------------- backward: g3 = Wo * d3 (reuse s_h) -------------
        {
            int m = nl, c0 = hf * 32;
            for (int t = 0; t < 32; ++t) {
                int c = c0 + t;
                float dd = s_d3[w][m * 64 + c] ? 1.0f : SLOPE;
                s_h[w][m * 64 + c] = (_Float16)(s_wo[c] * dd);
            }
        }

        // ---------------- g2 = (g3 @ W3) * d2 ----------------
        {
            v8f acc[4];
#pragma unroll
            for (int nt = 0; nt < 4; ++nt) {
                v8f c = {};
                for (int kb = 0; kb < 64; kb += 32)
                    c = wmma_f16(ldA(s_h[w], lane, kb), ldB(s_w3t, lane, nt, kb), c);
                acc[nt] = c;
            }
#pragma unroll
            for (int nt = 0; nt < 4; ++nt)
#pragma unroll
                for (int r = 0; r < 8; ++r) {
                    int m = r + hf * 8, c = nt * 16 + nl;
                    float dd = s_d2[w][m * 64 + c] ? 1.0f : SLOPE;
                    s_h[w][m * 64 + c] = (_Float16)(acc[nt][r] * dd);
                }
        }

        // ---------------- g1 = (g2 @ W2) * d1 ----------------
        {
            v8f acc[4];
#pragma unroll
            for (int nt = 0; nt < 4; ++nt) {
                v8f c = {};
                for (int kb = 0; kb < 64; kb += 32)
                    c = wmma_f16(ldA(s_h[w], lane, kb), ldB(s_w2t, lane, nt, kb), c);
                acc[nt] = c;
            }
#pragma unroll
            for (int nt = 0; nt < 4; ++nt)
#pragma unroll
                for (int r = 0; r < 8; ++r) {
                    int m = r + hf * 8, c = nt * 16 + nl;
                    float dd = s_d1[w][m * 64 + c] ? 1.0f : SLOPE;
                    s_h[w][m * 64 + c] = (_Float16)(acc[nt][r] * dd);
                }
        }

        // ---------------- J = g1 @ W1 (16 x fin, zero-padded N tile) -----
        {
            v8f cj = {};
            for (int kb = 0; kb < 64; kb += 32)
                cj = wmma_f16(ldA(s_h[w], lane, kb), ldB(s_w1t, lane, 0, kb), cj);

            int n = nl;
            if (n < fin - 1) {
#pragma unroll
                for (int r = 0; r < 8; ++r) {
                    int m = r + hf * 8;
                    hist_out[(size_t)(tile + m) * (fin - 1) + n] = cj[r];
                }
            } else if (n == fin - 1) {
#pragma unroll
                for (int r = 0; r < 8; ++r)
                    logdet_part += logf(fabsf(cj[r]));
            }
        }
    }

    // two atomics per wave per block (one per half-wave) (all ITERS tiles share the batch row)
    if (nl == fin - 1) {
        __hip_atomic_fetch_add(&logdet_out[base >> 9], logdet_part,
                               __ATOMIC_RELAXED, __HIP_MEMORY_SCOPE_AGENT);
    }
}

// ---------------- host side ----------------
extern "C" void kernel_launch(void* const* d_in, const int* in_sizes, int n_in,
                              void* d_out, int out_size, void* d_ws, size_t ws_size,
                              hipStream_t stream) {
    (void)in_sizes; (void)n_in; (void)out_size; (void)d_ws; (void)ws_size;

    static const int  FIN[7]   = {2, 4, 4, 7, 7, 7, 7};
    static const int  NYY[7]   = {1, 2, 2, 4, 4, 4, 4};
    static const int  YYOFF[7] = {0, 1, 1, 3, 3, 3, 3};
    static const int  NXP[7]   = {0, 1, 1, 2, 2, 2, 2};
    static const int  XPOFF[7] = {0, 0, 0, 1, 1, 1, 1};
    static const long HBASE[7] = {0, 1, 4, 7, 13, 19, 25}; // * N floats

    const float* x = (const float*)d_in[0];
    float* out    = (float*)d_out;
    float* res    = out;                           // N*7
    float* logdet = out + (size_t)NSAMP * DLAT;    // 256
    float* hist0  = logdet + 256;                  // N*31

    zero_logdet_kernel<<<1, 256, 0, stream>>>(logdet);

    for (int i = 0; i < 7; ++i) {
        const float* W1 = (const float*)d_in[1 + i * 8 + 0];
        const float* B1 = (const float*)d_in[1 + i * 8 + 1];
        const float* W2 = (const float*)d_in[1 + i * 8 + 2];
        const float* B2 = (const float*)d_in[1 + i * 8 + 3];
        const float* W3 = (const float*)d_in[1 + i * 8 + 4];
        const float* B3 = (const float*)d_in[1 + i * 8 + 5];
        const float* Wo = (const float*)d_in[1 + i * 8 + 6];
        const float* Bo = (const float*)d_in[1 + i * 8 + 7];
        float* hist = hist0 + (size_t)HBASE[i] * NSAMP;

        prior_mlp_kernel<<<NSAMP / (ITERS * 64), 128, 0, stream>>>(
            x, W1, B1, W2, B2, W3, B3, Wo, Bo, res, logdet, hist,
            FIN[i], NYY[i], YYOFF[i], NXP[i], XPOFF[i], /*self_col=*/i, i);
    }
}
